// Repro_10256381903435
// MI455X (gfx1250) — compile-verified, matrix operand order checked by reading, not run on hardware
//
#include <hip/hip_runtime.h>
#include <hip/hip_bf16.h>

// ---------------------------------------------------------------------------
// Chunked large-vocab softmax cross-entropy with tanh soft-capping (fp32 ref)
//   z    = 30 * tanh((x @ W^T) / 30)
//   loss = mean_valid( logsumexp_V(z) - z[label] )
// Strategy (MI455X / gfx1250, wave32, WMMA):
//   * GEMM in bf16 WMMA (16x16x32) with 3-term hi/lo split (x=xh+xl, w=wh+wl,
//     z ~= xh*wh + xh*wl + xl*wh) -> ~2^-16 relative logit error, near fp32.
//   * tanh cap bounds |z| <= 30 => sum(exp(z)) <= 262400*e^30 << fp32 max =>
//     single streaming pass, no row-max; weight (1.07 GB) read exactly once.
//   * Register double-buffering: next K-slab's global_load_b128s are issued
//     before the current slab's 24 WMMAs, hiding HBM latency.
//   * Per-block partial exp-sums reduced via shfl + LDS ds_add_f32, then one
//     global atomic per row per block.
// ---------------------------------------------------------------------------

#define NTOK   576
#define DDIM   1024
#define VOCAB  262400
#define BM     64
#define BN     128
#define BK     64
#define LDS_K  72          // BK + 8 pad; row pitch 144 B keeps 16-B alignment
#define SOFTCAPF 30.0f
#define IGNORE_INDEX (-100)

typedef __attribute__((ext_vector_type(16))) __bf16 v16bf;
typedef __attribute__((ext_vector_type(8)))  __bf16 v8bf;
typedef __attribute__((ext_vector_type(4)))  __bf16 v4bf;
typedef __attribute__((ext_vector_type(8)))  float  v8f;

// Load one 16x32 bf16 fragment (A-style layout; B uses the same layout since
// B = (W tile)^T and W rows are K-contiguous, mirroring A):
//   lanes 0-15 : row = lane,    K = [kk+0..7]  and [kk+16..23]
//   lanes 16-31: row = lane-16, K = [kk+8..15] and [kk+24..31]
__device__ __forceinline__ v16bf frag_ld(const __bf16* base, int row, int kk, int lane) {
    const __bf16* p = base + row * LDS_K + kk + ((lane >> 4) << 3);
    v8bf lo = *(const v8bf*)(p);
    v8bf hi = *(const v8bf*)(p + 16);
    return __builtin_shufflevector(lo, hi, 0, 1, 2, 3, 4, 5, 6, 7,
                                           8, 9, 10, 11, 12, 13, 14, 15);
}

// Convert 4 fp32 -> bf16 hi/lo planes, one 8-byte ds_store per plane.
__device__ __forceinline__ void cvt_store(__bf16* hp, __bf16* lp, float4 f) {
    float v[4] = {f.x, f.y, f.z, f.w};
    v4bf h, l;
    #pragma unroll
    for (int j = 0; j < 4; ++j) {
        __bf16 hv = (__bf16)v[j];
        h[j] = hv;
        l[j] = (__bf16)(v[j] - (float)hv);
    }
    *(v4bf*)hp = h;
    *(v4bf*)lp = l;
}

__global__ void cel_init(float* __restrict__ sumexp, float* __restrict__ tgt) {
    int i = blockIdx.x * blockDim.x + threadIdx.x;
    if (i < NTOK) {
        sumexp[i] = 0.0f;
        tgt[i]    = 0.0f;
    }
}

__global__ __launch_bounds__(256)
void cel_gemm(const float* __restrict__ x, const float* __restrict__ w,
              const long long* __restrict__ labels,
              float* __restrict__ sumexp_g, float* __restrict__ tgt_g) {
    __shared__ __align__(16) __bf16 xh[BM][LDS_K];
    __shared__ __align__(16) __bf16 xl[BM][LDS_K];
    __shared__ __align__(16) __bf16 wh[BN][LDS_K];
    __shared__ __align__(16) __bf16 wl[BN][LDS_K];
    __shared__ float rowsum[BM];

    const int tid  = threadIdx.x;
    const int lane = tid & 31;
    const int wave = tid >> 5;
    const int wm   = wave >> 2;                 // 0..1  -> M offset 32*wm
    const int wn   = wave & 3;                  // 0..3  -> N offset 32*wn
    const int mBase = blockIdx.x * BM;          // 9 blocks cover N=576
    const int vBase = blockIdx.y * BN;          // 2050 blocks cover V=262400

    if (tid < BM) rowsum[tid] = 0.0f;

    const v8f vzero = {0.f, 0.f, 0.f, 0.f, 0.f, 0.f, 0.f, 0.f};
    v8f acc[2][2];
    #pragma unroll
    for (int a = 0; a < 2; ++a)
        #pragma unroll
        for (int b = 0; b < 2; ++b)
            acc[a][b] = vzero;

    // ---- register double-buffer: prologue loads for k0 = 0 ----------------
    float4 sx[4], sw[8];
    #pragma unroll
    for (int i = 0; i < 4; ++i) {
        int idx = tid + i * 256;                // 0..1023 float4 slots
        int r   = idx >> 4;                     // 16 float4 per row
        int c   = (idx & 15) << 2;
        sx[i] = *(const float4*)(x + (size_t)(mBase + r) * DDIM + c);
    }
    #pragma unroll
    for (int i = 0; i < 8; ++i) {
        int idx = tid + i * 256;                // 0..2047 float4 slots
        int r   = idx >> 4;
        int c   = (idx & 15) << 2;
        sw[i] = *(const float4*)(w + (size_t)(vBase + r) * DDIM + c);
    }

    for (int k0 = 0; k0 < DDIM; k0 += BK) {
        // ---- drain staged registers into LDS hi/lo planes -----------------
        #pragma unroll
        for (int i = 0; i < 4; ++i) {
            int idx = tid + i * 256;
            int r   = idx >> 4;
            int c   = (idx & 15) << 2;
            cvt_store(&xh[r][c], &xl[r][c], sx[i]);
        }
        #pragma unroll
        for (int i = 0; i < 8; ++i) {
            int idx = tid + i * 256;
            int r   = idx >> 4;
            int c   = (idx & 15) << 2;
            cvt_store(&wh[r][c], &wl[r][c], sw[i]);
        }
        __syncthreads();

        // ---- issue next slab's global loads BEFORE the WMMA block ---------
        if (k0 + BK < DDIM) {
            #pragma unroll
            for (int i = 0; i < 4; ++i) {
                int idx = tid + i * 256;
                int r   = idx >> 4;
                int c   = (idx & 15) << 2;
                sx[i] = *(const float4*)(x + (size_t)(mBase + r) * DDIM + (k0 + BK) + c);
            }
            #pragma unroll
            for (int i = 0; i < 8; ++i) {
                int idx = tid + i * 256;
                int r   = idx >> 4;
                int c   = (idx & 15) << 2;
                sw[i] = *(const float4*)(w + (size_t)(vBase + r) * DDIM + (k0 + BK) + c);
            }
            if (k0 + 2 * BK < DDIM && tid < BN)
                __builtin_prefetch(w + (size_t)(vBase + tid) * DDIM + k0 + 2 * BK, 0, 1);
        }

        // ---- compute: 12 WMMAs per 32-wide K step (bf16 x3 split) ---------
        #pragma unroll
        for (int kk = 0; kk < BK; kk += 32) {
            v16bf Ah[2], Al[2], Bh[2], Bl[2];
            #pragma unroll
            for (int mt = 0; mt < 2; ++mt) {
                int r = wm * 32 + mt * 16 + (lane & 15);
                Ah[mt] = frag_ld(&xh[0][0], r, kk, lane);
                Al[mt] = frag_ld(&xl[0][0], r, kk, lane);
            }
            #pragma unroll
            for (int nt = 0; nt < 2; ++nt) {
                int r = wn * 32 + nt * 16 + (lane & 15);
                Bh[nt] = frag_ld(&wh[0][0], r, kk, lane);
                Bl[nt] = frag_ld(&wl[0][0], r, kk, lane);
            }
            #pragma unroll
            for (int mt = 0; mt < 2; ++mt)
                #pragma unroll
                for (int nt = 0; nt < 2; ++nt) {
                    acc[mt][nt] = __builtin_amdgcn_wmma_f32_16x16x32_bf16(
                        false, Ah[mt], false, Bh[nt], (short)0, acc[mt][nt], false, false);
                    acc[mt][nt] = __builtin_amdgcn_wmma_f32_16x16x32_bf16(
                        false, Ah[mt], false, Bl[nt], (short)0, acc[mt][nt], false, false);
                    acc[mt][nt] = __builtin_amdgcn_wmma_f32_16x16x32_bf16(
                        false, Al[mt], false, Bh[nt], (short)0, acc[mt][nt], false, false);
                }
        }
        __syncthreads();
    }

    // ---- epilogue: tanh-cap, exp, row reduction, target-logit scatter -----
    // C/D layout: lane l holds column N = (l & 15); VGPR j holds row
    // M = j + (l < 16 ? 0 : 8).
    const int half8 = (lane >> 4) << 3;
    #pragma unroll
    for (int mt = 0; mt < 2; ++mt) {
        #pragma unroll
        for (int j = 0; j < 8; ++j) {
            const int mloc = wm * 32 + mt * 16 + half8 + j;
            const int gm   = mBase + mloc;
            const int lbl  = (int)labels[gm];
            float esum = 0.0f;
            #pragma unroll
            for (int nt = 0; nt < 2; ++nt) {
                float z = SOFTCAPF * tanhf(acc[mt][nt][j] * (1.0f / SOFTCAPF));
                const int gv = vBase + wn * 32 + nt * 16 + (lane & 15);
                if (gv == lbl) tgt_g[gm] = z;   // unique writer per row
                esum += expf(z);
            }
            // reduce across the 16 lanes holding the same row (masks < 16
            // keep the two half-wave row groups separate)
            esum += __shfl_xor(esum, 1);
            esum += __shfl_xor(esum, 2);
            esum += __shfl_xor(esum, 4);
            esum += __shfl_xor(esum, 8);
            if ((lane & 15) == 0) atomicAdd(&rowsum[mloc], esum);
        }
    }
    __syncthreads();
    if (tid < BM) atomicAdd(&sumexp_g[mBase + tid], rowsum[tid]);
}

__global__ __launch_bounds__(256)
void cel_finalize(const float* __restrict__ sumexp, const float* __restrict__ tgt,
                  const long long* __restrict__ labels, float* __restrict__ out) {
    __shared__ float sred[256];
    __shared__ int   cred[256];
    float s = 0.0f;
    int   c = 0;
    for (int r = threadIdx.x; r < NTOK; r += 256) {
        long long lbl = labels[r];
        if (lbl != (long long)IGNORE_INDEX) {
            s += logf(sumexp[r]) - tgt[r];
            c += 1;
        }
    }
    sred[threadIdx.x] = s;
    cred[threadIdx.x] = c;
    __syncthreads();
    for (int off = 128; off > 0; off >>= 1) {
        if (threadIdx.x < off) {
            sred[threadIdx.x] += sred[threadIdx.x + off];
            cred[threadIdx.x] += cred[threadIdx.x + off];
        }
        __syncthreads();
    }
    if (threadIdx.x == 0) out[0] = sred[0] / (float)cred[0];
}

extern "C" void kernel_launch(void* const* d_in, const int* in_sizes, int n_in,
                              void* d_out, int out_size, void* d_ws, size_t ws_size,
                              hipStream_t stream) {
    const float*     x      = (const float*)d_in[0];       // [576, 1024] fp32
    const float*     w      = (const float*)d_in[1];       // [262400, 1024] fp32
    const long long* labels = (const long long*)d_in[2];   // [576] int64
    float* out = (float*)d_out;                            // scalar fp32

    float* sumexp = (float*)d_ws;                          // [576]
    float* tgt    = sumexp + NTOK;                         // [576]

    cel_init<<<(NTOK + 255) / 256, 256, 0, stream>>>(sumexp, tgt);

    dim3 grid(NTOK / BM, VOCAB / BN);                      // (9, 2050)
    cel_gemm<<<grid, 256, 0, stream>>>(x, w, labels, sumexp, tgt);

    cel_finalize<<<1, 256, 0, stream>>>(sumexp, tgt, labels, out);
}
